// ScaledDotProductAttention_87522843558151
// MI455X (gfx1250) — matrix operand order repaired and checked by reading, not compiled, not run on hardware
//
#include <hip/hip_runtime.h>
#include <cstdint>

// Problem constants (B, H, S, DK) = (2, 16, 2048, 64)
#define BATCH   2
#define HEADS   16
#define S_LEN   2048
#define DKDIM   64

#define BQ      128           // queries per workgroup (4 waves x 2 M-tiles x 16 rows)
#define BK      64            // keys per iteration
#define NW      4             // waves per workgroup
#define M2      2             // M-tiles (16 rows each) per wave
#define LSTR    72            // LDS row stride in bf16 elements (pad vs bank conflicts)

typedef __bf16 bf16_t;
typedef __attribute__((ext_vector_type(16))) __bf16    v16bf;
typedef __attribute__((ext_vector_type(8)))  float     v8f;
typedef __attribute__((ext_vector_type(4)))  float     v4f;
typedef __attribute__((ext_vector_type(4)))  uint32_t  v4u;

// 32-byte fragment viewed either as the WMMA bf16 operand or as two 128-bit
// LDS load payloads.
union FragA {
    v16bf v;
    v4u   q[2];
};

__global__ __launch_bounds__(NW * 32) void
fa_fwd_bf16_wmma(const float* __restrict__ Q,
                 const float* __restrict__ K,
                 const float* __restrict__ V,
                 float* __restrict__ O)
{
    __shared__ __align__(16) bf16_t q_s[BQ * LSTR];          // [query][dk] (bf16, pre-scaled)
    __shared__ __align__(16) bf16_t k_s[BK * LSTR];          // [key][dk]
    __shared__ __align__(16) bf16_t v_s[DKDIM * LSTR];       // [dk][key]   (transposed)
    __shared__ __align__(16) bf16_t p_s[NW * 32 * LSTR];     // per-wave P staging [row][key]

    const int tid  = threadIdx.x;
    const int lane = tid & 31;
    const int wave = tid >> 5;
    const int hi   = (lane >> 4) & 1;   // half-wave selector (CDNA5 wave32 layouts)
    const int ln   = lane & 15;

    const int qt = blockIdx.x;          // query tile index   (S/BQ = 16)
    const int h  = blockIdx.y;          // head               (16)
    const int b  = blockIdx.z;          // batch              (2)

    const size_t head_off = ((size_t)(b * HEADS + h)) * S_LEN * DKDIM;
    const float* Qp = Q + head_off + (size_t)qt * BQ * DKDIM;
    const float* Kp = K + head_off;
    const float* Vp = V + head_off;

    // ---- stage Q tile (128x64) into LDS as bf16, 1/sqrt(dk)=0.125 folded in ----
    {
        const int r0 = tid >> 4;          // 0..7
        const int c0 = (tid & 15) * 4;    // 0,4,...,60
        #pragma unroll
        for (int r = 0; r < 16; ++r) {
            const int row = r0 + r * 8;   // 0..127
            v4f f = *(const v4f*)(Qp + row * DKDIM + c0);
            bf16_t* dst = &q_s[row * LSTR + c0];
            dst[0] = (bf16_t)(f.x * 0.125f);
            dst[1] = (bf16_t)(f.y * 0.125f);
            dst[2] = (bf16_t)(f.z * 0.125f);
            dst[3] = (bf16_t)(f.w * 0.125f);
        }
    }
    __syncthreads();

    // ---- build Q A-fragments: [M-tile][DK chunk of 32] ----
    // A 16x32 bf16 layout: lane<16 -> row=ln, K = {0..7, 16..23};
    //                      lane>=16 -> row=ln, K = {8..15, 24..31}.
    FragA aq[M2][2];
    #pragma unroll
    for (int mt = 0; mt < M2; ++mt) {
        const bf16_t* src = &q_s[(wave * 32 + mt * 16 + ln) * LSTR];
        #pragma unroll
        for (int c = 0; c < 2; ++c) {
            const int kb0 = c * 32 + hi * 8;
            aq[mt][c].q[0] = *(const v4u*)(src + kb0);
            aq[mt][c].q[1] = *(const v4u*)(src + kb0 + 16);
        }
    }

    // ---- flash-attention running state (8 C-rows per lane per M-tile) ----
    float m_i[M2][8], l_i[M2][8];
    v8f   o_acc[M2][4];
    const v8f zacc = {0.f, 0.f, 0.f, 0.f, 0.f, 0.f, 0.f, 0.f};
    #pragma unroll
    for (int mt = 0; mt < M2; ++mt) {
        #pragma unroll
        for (int i = 0; i < 8; ++i) { m_i[mt][i] = -1e30f; l_i[mt][i] = 0.f; }
        #pragma unroll
        for (int g = 0; g < 4; ++g) o_acc[mt][g] = zacc;
    }

    for (int kb = 0; kb < S_LEN / BK; ++kb) {
        __syncthreads();   // previous iteration's k_s/v_s reads are done

        const float* kp = Kp + (size_t)kb * BK * DKDIM;
        const float* vp = Vp + (size_t)kb * BK * DKDIM;

        // prefetch next K/V tiles into cache while we compute this one
        if (kb + 1 < S_LEN / BK) {
            __builtin_prefetch(kp + BK * DKDIM + tid * 32, 0, 3);
            __builtin_prefetch(vp + BK * DKDIM + tid * 32, 0, 3);
        }

        // ---- stage K (row-major) and V (transposed) tiles as bf16 ----
        {
            const int r0 = tid >> 4;
            const int c0 = (tid & 15) * 4;
            #pragma unroll
            for (int r = 0; r < 8; ++r) {
                const int row = r0 + r * 8;
                v4f f = *(const v4f*)(kp + row * DKDIM + c0);
                bf16_t* dst = &k_s[row * LSTR + c0];
                dst[0] = (bf16_t)f.x;
                dst[1] = (bf16_t)f.y;
                dst[2] = (bf16_t)f.z;
                dst[3] = (bf16_t)f.w;
                v4f g4 = *(const v4f*)(vp + row * DKDIM + c0);
                v_s[(c0 + 0) * LSTR + row] = (bf16_t)g4.x;
                v_s[(c0 + 1) * LSTR + row] = (bf16_t)g4.y;
                v_s[(c0 + 2) * LSTR + row] = (bf16_t)g4.z;
                v_s[(c0 + 3) * LSTR + row] = (bf16_t)g4.w;
            }
        }
        __syncthreads();

        // ---- scores: S(32x64) = Qtiles . K^T ----
        // Each B-fragment (bk) is loaded once and feeds both M-tiles' WMMAs.
        // B 32x16 bf16 layout: lane holds column N=ln, K = {0..7, 16..23} (+8 for hi),
        // base = chunk*32 + hi*8.
        v8f sc[M2][4];
        #pragma unroll
        for (int mt = 0; mt < M2; ++mt)
            #pragma unroll
            for (int g = 0; g < 4; ++g) sc[mt][g] = zacc;

        #pragma unroll
        for (int g = 0; g < 4; ++g) {
            #pragma unroll
            for (int cc = 0; cc < 2; ++cc) {
                FragA bk;
                const bf16_t* src = &k_s[(g * 16 + ln) * LSTR + cc * 32 + hi * 8];
                bk.q[0] = *(const v4u*)(src);
                bk.q[1] = *(const v4u*)(src + 16);
                #pragma unroll
                for (int mt = 0; mt < M2; ++mt)
                    sc[mt][g] = __builtin_amdgcn_wmma_f32_16x16x32_bf16(
                            false, aq[mt][cc].v, false, bk.v, (short)0,
                            sc[mt][g], false, false);
            }
        }

        // ---- online softmax + P restage, per M-tile ----
        #pragma unroll
        for (int mt = 0; mt < M2; ++mt) {
            #pragma unroll
            for (int i = 0; i < 8; ++i) {
                float mx = sc[mt][0][i];
                mx = fmaxf(mx, sc[mt][1][i]);
                mx = fmaxf(mx, sc[mt][2][i]);
                mx = fmaxf(mx, sc[mt][3][i]);
                mx = fmaxf(mx, __shfl_xor(mx, 1));
                mx = fmaxf(mx, __shfl_xor(mx, 2));
                mx = fmaxf(mx, __shfl_xor(mx, 4));
                mx = fmaxf(mx, __shfl_xor(mx, 8));
                const float mnew  = fmaxf(m_i[mt][i], mx);
                const float alpha = __expf(m_i[mt][i] - mnew);
                m_i[mt][i] = mnew;

                float rsum = 0.f;
                #pragma unroll
                for (int g = 0; g < 4; ++g) {
                    const float p = __expf(sc[mt][g][i] - mnew);
                    sc[mt][g][i] = p;
                    rsum += p;
                }
                rsum += __shfl_xor(rsum, 1);
                rsum += __shfl_xor(rsum, 2);
                rsum += __shfl_xor(rsum, 4);
                rsum += __shfl_xor(rsum, 8);
                l_i[mt][i] = l_i[mt][i] * alpha + rsum;

                #pragma unroll
                for (int g = 0; g < 4; ++g)
                    o_acc[mt][g][i] = o_acc[mt][g][i] * alpha;
            }

            // restage P as bf16 in this wave's private LDS region
            // (same-wave LDS store->load is in-order; no workgroup barrier)
            bf16_t* prow = &p_s[(wave * 32 + mt * 16) * LSTR];
            #pragma unroll
            for (int g = 0; g < 4; ++g) {
                #pragma unroll
                for (int i = 0; i < 8; ++i)
                    prow[(i + 8 * hi) * LSTR + g * 16 + ln] = (bf16_t)sc[mt][g][i];
            }
        }

        // ---- O(32x64) += P(32x64) . V(64x64) ----
        // Each V B-fragment (bv) is loaded once and feeds both M-tiles.
        #pragma unroll
        for (int cc = 0; cc < 2; ++cc) {
            FragA ap[M2];
            #pragma unroll
            for (int mt = 0; mt < M2; ++mt) {
                const bf16_t* ps =
                    &p_s[(wave * 32 + mt * 16 + ln) * LSTR + cc * 32 + hi * 8];
                ap[mt].q[0] = *(const v4u*)(ps);
                ap[mt].q[1] = *(const v4u*)(ps + 16);
            }
            #pragma unroll
            for (int g = 0; g < 4; ++g) {
                FragA bv;
                const bf16_t* vs = &v_s[(g * 16 + ln) * LSTR + cc * 32 + hi * 8];
                bv.q[0] = *(const v4u*)(vs);
                bv.q[1] = *(const v4u*)(vs + 16);
                #pragma unroll
                for (int mt = 0; mt < M2; ++mt)
                    o_acc[mt][g] = __builtin_amdgcn_wmma_f32_16x16x32_bf16(
                            false, ap[mt].v, false, bv.v, (short)0,
                            o_acc[mt][g], false, false);
            }
        }
    }

    // ---- epilogue: O /= l, scatter to [B, S, H*DK] ----
    {
        float* Ob = O + (size_t)b * S_LEN * (HEADS * DKDIM) + (size_t)h * DKDIM;
        #pragma unroll
        for (int mt = 0; mt < M2; ++mt) {
            const int qrow0 = qt * BQ + wave * 32 + mt * 16;
            #pragma unroll
            for (int i = 0; i < 8; ++i) {
                const float inv = 1.0f / l_i[mt][i];
                const int row = qrow0 + i + 8 * hi;
                #pragma unroll
                for (int g = 0; g < 4; ++g)
                    Ob[(size_t)row * (HEADS * DKDIM) + g * 16 + ln] =
                        o_acc[mt][g][i] * inv;
            }
        }
    }
}

extern "C" void kernel_launch(void* const* d_in, const int* in_sizes, int n_in,
                              void* d_out, int out_size, void* d_ws, size_t ws_size,
                              hipStream_t stream) {
    (void)in_sizes; (void)n_in; (void)out_size; (void)d_ws; (void)ws_size;
    const float* q = (const float*)d_in[0];
    const float* k = (const float*)d_in[1];
    const float* v = (const float*)d_in[2];
    float* o = (float*)d_out;

    dim3 grid(S_LEN / BQ, HEADS, BATCH);   // 16 x 16 x 2 = 512 workgroups
    dim3 block(NW * 32);                   // 128 threads = 4 wave32
    fa_fwd_bf16_wmma<<<grid, block, 0, stream>>>(q, k, v, o);
}
